// MoEPairwise_20255065768147
// MI455X (gfx1250) — compile-verified, matrix operand order checked by reading, not run on hardware
//
#include <hip/hip_runtime.h>
#include <hip/hip_bf16.h>

typedef __attribute__((ext_vector_type(16))) _Float16 v16h;
typedef __attribute__((ext_vector_type(8)))  _Float16 v8h;
typedef __attribute__((ext_vector_type(2)))  __fp16   v2fp;
typedef __attribute__((ext_vector_type(8)))  float    v8f;

union V16u { v16h v; v8h h8[2]; v2fp hp[8]; };

__device__ __forceinline__ float4 ld4(const float* p) { return *(const float4*)p; }

// pack 4 float4 (two contiguous 8-float runs) into one v16h via packed cvt
__device__ __forceinline__ v16h cvt16(float4 a0, float4 a1, float4 b0, float4 b1) {
  V16u u;
  u.hp[0] = __builtin_amdgcn_cvt_pkrtz(a0.x, a0.y);
  u.hp[1] = __builtin_amdgcn_cvt_pkrtz(a0.z, a0.w);
  u.hp[2] = __builtin_amdgcn_cvt_pkrtz(a1.x, a1.y);
  u.hp[3] = __builtin_amdgcn_cvt_pkrtz(a1.z, a1.w);
  u.hp[4] = __builtin_amdgcn_cvt_pkrtz(b0.x, b0.y);
  u.hp[5] = __builtin_amdgcn_cvt_pkrtz(b0.z, b0.w);
  u.hp[6] = __builtin_amdgcn_cvt_pkrtz(b1.x, b1.y);
  u.hp[7] = __builtin_amdgcn_cvt_pkrtz(b1.z, b1.w);
  return u.v;
}

// Unguarded A-tile loader (16-bit A-matrix 16x32 lane layout). Caller
// guarantees Arow[k0 .. k0+31] is valid (zero-padded where needed).
__device__ __forceinline__ v16h load_a_f32(const float* __restrict__ Arow,
                                           int k0, int half) {
  const float* p = Arow + k0 + half * 8;
  return cvt16(ld4(p), ld4(p + 4), ld4(p + 16), ld4(p + 20));
}

// A-tile from f16 row in LDS (two 16-byte ds_load_b128)
__device__ __forceinline__ v16h load_a_lds(const _Float16* __restrict__ arow,
                                           int k0, int half) {
  V16u u;
  u.h8[0] = *(const v8h*)(arow + k0 + half * 8);
  u.h8[1] = *(const v8h*)(arow + k0 + 16 + half * 8);
  return u.v;
}

__device__ __forceinline__ float4 f4sub(float4 a, float4 b) {
  return make_float4(a.x - b.x, a.y - b.y, a.z - b.z, a.w - b.w);
}
__device__ __forceinline__ float4 f4mul(float4 a, float4 b) {
  return make_float4(a.x * b.x, a.y * b.y, a.z * b.z, a.w * b.w);
}

// ---------------------------------------------------------------------------
// Weight packing into WMMA B-matrix lane layout (16-bit 32x16 tiles),
// zero-padded to (Kp, Np). tile = nt*(Kp/32)+kt; lane l: n = nt*16+(l&15),
// half = l>>4; t<8: k = kt*32+half*8+t ; t>=8: k = kt*32+16+half*8+(t-8).
// ---------------------------------------------------------------------------
__global__ void k_pack(const float* __restrict__ W, _Float16* __restrict__ Wp,
                       int K, int N, int Kp, int Np, int batch,
                       int sStride, int dStride) {
  int idx = blockIdx.x * blockDim.x + threadIdx.x;
  int per = Kp * Np;
  int e = idx / per;
  if (e >= batch) return;
  int p = idx - e * per;
  int tile = p >> 9;
  int within = p & 511;
  int lane = within >> 4;
  int t = within & 15;
  int KT = Kp >> 5;
  int nt = tile / KT, kt = tile - nt * KT;
  int half = lane >> 4;
  int n = nt * 16 + (lane & 15);
  int k = kt * 32 + ((t < 8) ? (half * 8 + t) : (16 + half * 8 + (t - 8)));
  float v = (k < K && n < N) ? W[(size_t)e * sStride + (size_t)k * N + n] : 0.0f;
  Wp[(size_t)e * dStride + p] = (_Float16)v;
}

// ---------------------------------------------------------------------------
// Generic WMMA GEMM: C = act(A[M,Kp] @ W[Kp,Np] + bias). One wave per block,
// NT 16x16 n-tiles per wave (A tile reused NT times). grid = (M/16, Np/(16*NT))
// ---------------------------------------------------------------------------
template <int NT>
__global__ void __launch_bounds__(32)
k_gemm(const float* __restrict__ A, int lda, int Kp,
       const _Float16* __restrict__ Wp, const float* __restrict__ bias,
       float* __restrict__ C, int ldc, int Nreal, int relu) {
  const int lane = threadIdx.x;
  const int half = lane >> 4;
  const int m0 = blockIdx.x * 16;
  const int nt0 = blockIdx.y * NT;
  const int KT = Kp >> 5;
  const v16h* __restrict__ Bt = (const v16h*)Wp;
  const float* __restrict__ Arow = A + (size_t)(m0 + (lane & 15)) * lda;
  v8f acc[NT] = {};
  for (int kt = 0; kt < KT; ++kt) {
    __builtin_prefetch((const void*)&Bt[((size_t)nt0 * KT + kt + 1) * 32 + lane], 0, 3);
    v16h a = load_a_f32(Arow, kt * 32, half);
#pragma unroll
    for (int q = 0; q < NT; ++q) {
      v16h b = Bt[((size_t)(nt0 + q) * KT + kt) * 32 + lane];
      acc[q] = __builtin_amdgcn_wmma_f32_16x16x32_f16(false, a, false, b,
                                                      (short)0, acc[q],
                                                      false, false);
    }
  }
#pragma unroll
  for (int q = 0; q < NT; ++q) {
    int n = (nt0 + q) * 16 + (lane & 15);
    if (n < Nreal) {
      float bz = bias[n];
#pragma unroll
      for (int v = 0; v < 8; ++v) {
        float val = acc[q][v] + bz;
        if (relu) val = fmaxf(val, 0.0f);
        C[(size_t)(m0 + v + 8 * half) * ldc + n] = val;
      }
    }
  }
}

// ---------------------------------------------------------------------------
// rt_W1 GEMM with on-the-fly pair features (K = 1024, 4 sections of 256):
// sect 0: h[b,i] ; 1: h[b,j] ; 2: h[b,i]-h[b,j] ; 3: h[b,i]*h[b,j]
// Section index = kt>>3 is wave-uniform (scalar branch, EXEC untouched).
// 4 n-tiles per wave. grid = (512, 4).
// ---------------------------------------------------------------------------
__global__ void __launch_bounds__(32)
k_gemm_pair4(const float* __restrict__ h, const _Float16* __restrict__ Wp,
             const float* __restrict__ bias, float* __restrict__ C, int ldc) {
  const int lane = threadIdx.x;
  const int half = lane >> 4;
  const int m0 = blockIdx.x * 16;
  const int nt0 = blockIdx.y * 4;
  const int KT = 32; // Kp = 1024
  const v16h* __restrict__ Bt = (const v16h*)Wp;
  const int r = m0 + (lane & 15);
  const int b = r >> 10, i = (r >> 5) & 31, j = r & 31;
  const float* __restrict__ Hi = h + (size_t)((b << 5) + i) * 256;
  const float* __restrict__ Hj = h + (size_t)((b << 5) + j) * 256;
  v8f acc[4] = {};
  for (int kt = 0; kt < KT; ++kt) {
    const int sect = kt >> 3;
    const int d0 = ((kt & 7) << 5) + half * 8;
    float4 r0, r1, r2, r3;
    if (sect == 0) {
      r0 = ld4(Hi + d0); r1 = ld4(Hi + d0 + 4);
      r2 = ld4(Hi + d0 + 16); r3 = ld4(Hi + d0 + 20);
    } else if (sect == 1) {
      r0 = ld4(Hj + d0); r1 = ld4(Hj + d0 + 4);
      r2 = ld4(Hj + d0 + 16); r3 = ld4(Hj + d0 + 20);
    } else {
      float4 x0 = ld4(Hi + d0), x1 = ld4(Hi + d0 + 4);
      float4 x2 = ld4(Hi + d0 + 16), x3 = ld4(Hi + d0 + 20);
      float4 y0 = ld4(Hj + d0), y1 = ld4(Hj + d0 + 4);
      float4 y2 = ld4(Hj + d0 + 16), y3 = ld4(Hj + d0 + 20);
      if (sect == 2) {
        r0 = f4sub(x0, y0); r1 = f4sub(x1, y1);
        r2 = f4sub(x2, y2); r3 = f4sub(x3, y3);
      } else {
        r0 = f4mul(x0, y0); r1 = f4mul(x1, y1);
        r2 = f4mul(x2, y2); r3 = f4mul(x3, y3);
      }
    }
    v16h a = cvt16(r0, r1, r2, r3);
#pragma unroll
    for (int q = 0; q < 4; ++q) {
      v16h bb = Bt[((size_t)(nt0 + q) * KT + kt) * 32 + lane];
      acc[q] = __builtin_amdgcn_wmma_f32_16x16x32_f16(false, a, false, bb,
                                                      (short)0, acc[q],
                                                      false, false);
    }
  }
#pragma unroll
  for (int q = 0; q < 4; ++q) {
    int n = (nt0 + q) * 16 + (lane & 15);
    float bz = bias[n];
#pragma unroll
    for (int v = 0; v < 8; ++v)
      C[(size_t)(m0 + v + 8 * half) * ldc + n] = fmaxf(acc[q][v] + bz, 0.0f);
  }
}

// ---------------------------------------------------------------------------
// Fused expert stage. One block = 16 pair rows (b,i fixed; j = j0..j0+15).
// 4 waves; wave wv handles experts wv and wv+4 sequentially:
//   h1  = relu(pair_in @ exW1[e] + b1[e])   (f16, staged in LDS)
//   eff = h1 @ exW2[e] + b2[e]
//   VGPR wacc += weights[r,e] * eff         (accumulated across both experts)
// wacc -> LDS atomics (cross-wave sum over e) -> global atomics (sum over i).
// ---------------------------------------------------------------------------
__global__ void __launch_bounds__(128)
k_experts(const float* __restrict__ h,
          const _Float16* __restrict__ exW1p, const float* __restrict__ exb1,
          const _Float16* __restrict__ exW2p, const float* __restrict__ exb2,
          const float* __restrict__ wts,
          float* __restrict__ dinter, int ldc) {
  __shared__ _Float16 sh1[4][16 * 256];
  __shared__ float sacc[16 * 256];
  const int tid = threadIdx.x;
  const int lane = tid & 31, wv = tid >> 5;
  const int half = lane >> 4;
  const int r0 = blockIdx.x * 16;

  for (int p = tid; p < 16 * 256; p += 128) sacc[p] = 0.0f;
  __syncthreads();

  const int rm = r0 + (lane & 15);
  const int b = rm >> 10, i = (rm >> 5) & 31, j = rm & 31;
  const float* __restrict__ Hi = h + (size_t)((b << 5) + i) * 256;
  const float* __restrict__ Hj = h + (size_t)((b << 5) + j) * 256;
  _Float16* myh1 = sh1[wv];

  v8f wacc[16] = {}; // [nc*4+q] weighted expert output, both experts

  for (int ep = 0; ep < 2; ++ep) {
    const int e = wv + ep * 4;
    const v16h* __restrict__ B1 = (const v16h*)(exW1p + (size_t)e * 512 * 256);
    const v16h* __restrict__ B2 = (const v16h*)(exW2p + (size_t)e * 256 * 256);
    const float* __restrict__ b1 = exb1 + e * 256;
    const float* __restrict__ b2 = exb2 + e * 256;

    // GEMM1: pair_in(16x512) @ W1(512x256); sections: kt<8 -> Hi, else Hj
    for (int nc = 0; nc < 4; ++nc) {
      v8f acc[4] = {};
      for (int kt = 0; kt < 16; ++kt) {
        const float* __restrict__ src = (kt < 8) ? Hi : Hj;
        const int d0 = ((kt & 7) << 5) + half * 8;
        v16h a = cvt16(ld4(src + d0), ld4(src + d0 + 4),
                       ld4(src + d0 + 16), ld4(src + d0 + 20));
#pragma unroll
        for (int q = 0; q < 4; ++q) {
          int nt = nc * 4 + q;
          v16h bb = B1[(size_t)(nt * 16 + kt) * 32 + lane];
          acc[q] = __builtin_amdgcn_wmma_f32_16x16x32_f16(false, a, false, bb,
                                                          (short)0, acc[q],
                                                          false, false);
        }
      }
#pragma unroll
      for (int q = 0; q < 4; ++q) {
        int n = (nc * 4 + q) * 16 + (lane & 15);
#pragma unroll
        for (int v = 0; v < 8; ++v) {
          float val = fmaxf(acc[q][v] + b1[n], 0.0f);
          myh1[(v + 8 * half) * 256 + n] = (_Float16)val;
        }
      }
    }

    // GEMM2: h1(16x256 f16, LDS) @ W2(256x256); weighted accumulate in VGPRs
    const _Float16* __restrict__ arow = myh1 + (lane & 15) * 256;
    for (int nc = 0; nc < 4; ++nc) {
      v8f acc[4] = {};
      for (int kt = 0; kt < 8; ++kt) {
        v16h a = load_a_lds(arow, kt * 32, half);
#pragma unroll
        for (int q = 0; q < 4; ++q) {
          int nt = nc * 4 + q;
          v16h bb = B2[(size_t)(nt * 8 + kt) * 32 + lane];
          acc[q] = __builtin_amdgcn_wmma_f32_16x16x32_f16(false, a, false, bb,
                                                          (short)0, acc[q],
                                                          false, false);
        }
      }
#pragma unroll
      for (int q = 0; q < 4; ++q) {
        int n = (nc * 4 + q) * 16 + (lane & 15);
#pragma unroll
        for (int v = 0; v < 8; ++v) {
          int ml = v + 8 * half;
          float wgt = wts[(size_t)(r0 + ml) * 8 + e];
          wacc[nc * 4 + q][v] += (acc[q][v] + b2[n]) * wgt;
        }
      }
    }
  }

  // cross-wave reduction (sum over experts) via LDS atomics
#pragma unroll
  for (int nq = 0; nq < 16; ++nq) {
    int n = nq * 16 + (lane & 15);
#pragma unroll
    for (int v = 0; v < 8; ++v)
      atomicAdd(&sacc[(v + 8 * half) * 256 + n], wacc[nq][v]);
  }
  __syncthreads();

  // sum over i across blocks
  for (int p = tid; p < 16 * 256; p += 128) {
    int ml = p >> 8, n = p & 255;
    int rr = r0 + ml;
    int bb2 = rr >> 10, jj = rr & 31;
    atomicAdd(&dinter[(size_t)((bb2 << 5) + jj) * ldc + n], sacc[p]);
  }
}

// ---------------------------------------------------------------------------
// LayerNorm over last dim: one block (256 threads) per row.
// ---------------------------------------------------------------------------
__global__ void __launch_bounds__(256)
k_ln(const float* __restrict__ X, const float* __restrict__ G,
     const float* __restrict__ Bt, float* __restrict__ Y, int D) {
  __shared__ float rs[256], rs2[256];
  const int row = blockIdx.x, tid = threadIdx.x;
  const float* x = X + (size_t)row * D;
  float s = 0.0f, s2 = 0.0f;
  for (int d = tid; d < D; d += 256) { float v = x[d]; s += v; s2 += v * v; }
  rs[tid] = s; rs2[tid] = s2;
  __syncthreads();
  for (int off = 128; off > 0; off >>= 1) {
    if (tid < off) { rs[tid] += rs[tid + off]; rs2[tid] += rs2[tid + off]; }
    __syncthreads();
  }
  float mean = rs[0] / (float)D;
  float var = rs2[0] / (float)D - mean * mean;
  float inv = rsqrtf(var + 1e-5f);
  float* y = Y + (size_t)row * D;
  for (int d = tid; d < D; d += 256)
    y[d] = (x[d] - mean) * inv * G[d] + Bt[d];
}

__global__ void k_softmax8(const float* __restrict__ L, float* __restrict__ W,
                           int R) {
  int r = blockIdx.x * blockDim.x + threadIdx.x;
  if (r >= R) return;
  const float* l = L + (size_t)r * 16;
  float mx = l[0];
#pragma unroll
  for (int e = 1; e < 8; ++e) mx = fmaxf(mx, l[e]);
  float ex[8], s = 0.0f;
#pragma unroll
  for (int e = 0; e < 8; ++e) { ex[e] = __expf(l[e] - mx); s += ex[e]; }
  float inv = 1.0f / s;
#pragma unroll
  for (int e = 0; e < 8; ++e) W[(size_t)r * 8 + e] = ex[e] * inv;
}

__global__ void k_zero(float* __restrict__ p, int n) {
  int i = blockIdx.x * blockDim.x + threadIdx.x;
  if (i < n) p[i] = 0.0f;
}

// gt_states (B=8,T=3,K=32,Ds=8) t=0 -> x (256 x 32, zero-padded)
__global__ void k_gather_x0(const float* __restrict__ gt, float* __restrict__ x) {
  int idx = blockIdx.x * blockDim.x + threadIdx.x;
  if (idx < 2048)
    x[(size_t)(idx >> 3) * 32 + (idx & 7)] =
        gt[(size_t)(idx >> 8) * 768 + (idx & 255)];
}

// pred (256 x 32, cols 0..7 valid) -> d_out[:4096] step s : (B,2,K,Ds)
__global__ void k_scatter_pred(const float* __restrict__ pred,
                               float* __restrict__ out, int s) {
  int idx = blockIdx.x * blockDim.x + threadIdx.x;
  if (idx < 2048)
    out[(size_t)(idx >> 8) * 512 + s * 256 + (idx & 255)] =
        pred[(size_t)(idx >> 3) * 32 + (idx & 7)];
}

// gt_states[:,1:3] -> d_out[4096:8192]
__global__ void k_copy_targets(const float* __restrict__ gt,
                               float* __restrict__ out) {
  int idx = blockIdx.x * blockDim.x + threadIdx.x;
  if (idx < 4096)
    out[idx] = gt[(size_t)(idx >> 9) * 768 + 256 + (idx & 511)];
}

// ---------------------------------------------------------------------------
extern "C" void kernel_launch(void* const* d_in, const int* in_sizes, int n_in,
                              void* d_out, int out_size, void* d_ws,
                              size_t ws_size, hipStream_t stream) {
  const float* gt     = (const float*)d_in[0];
  const float* enc_W1 = (const float*)d_in[1];
  const float* enc_b1 = (const float*)d_in[2];
  const float* enc_W2 = (const float*)d_in[3];
  const float* enc_b2 = (const float*)d_in[4];
  const float* rt_W1  = (const float*)d_in[5];
  const float* rt_b1  = (const float*)d_in[6];
  const float* rt_W2  = (const float*)d_in[7];
  const float* rt_b2  = (const float*)d_in[8];
  const float* fs_g   = (const float*)d_in[9];
  const float* fs_b   = (const float*)d_in[10];
  const float* fs_W1  = (const float*)d_in[11];
  const float* fs_b1  = (const float*)d_in[12];
  const float* fs_W2  = (const float*)d_in[13];
  const float* fs_b2  = (const float*)d_in[14];
  const float* ex_W1  = (const float*)d_in[15];
  const float* ex_b1  = (const float*)d_in[16];
  const float* ex_W2  = (const float*)d_in[17];
  const float* ex_b2  = (const float*)d_in[18];
  const float* up_g   = (const float*)d_in[19];
  const float* up_b   = (const float*)d_in[20];
  const float* up_W1  = (const float*)d_in[21];
  const float* up_b1  = (const float*)d_in[22];
  const float* up_W2  = (const float*)d_in[23];
  const float* up_b2  = (const float*)d_in[24];
  const float* dec_W1 = (const float*)d_in[25];
  const float* dec_b1 = (const float*)d_in[26];
  const float* dec_W2 = (const float*)d_in[27];
  const float* dec_b2 = (const float*)d_in[28];
  // d_in[29] = rollout_steps (== 2 from setup; min(T-1,2)=2, unrolled below)
  (void)in_sizes; (void)n_in; (void)out_size; (void)ws_size;

  char* base = (char*)d_ws;
  size_t off = 0;
  auto A16 = [&](size_t n) {
    void* p = base + off; off += (n * 2 + 255) & ~(size_t)255; return (_Float16*)p;
  };
  auto AF = [&](size_t n) {
    void* p = base + off; off += (n * 4 + 255) & ~(size_t)255; return (float*)p;
  };
  _Float16* encW1p = A16(32 * 256);
  _Float16* encW2p = A16(256 * 256);
  _Float16* rtW1p  = A16(1024 * 256);
  _Float16* rtW2p  = A16(256 * 16);
  _Float16* fsW1p  = A16(256 * 256);
  _Float16* fsW2p  = A16(256 * 256);
  _Float16* exW1p  = A16(8 * 512 * 256);
  _Float16* exW2p  = A16(8 * 256 * 256);
  _Float16* upW1p  = A16(512 * 256);
  _Float16* upW2p  = A16(256 * 256);
  _Float16* decW1p = A16(256 * 256);
  _Float16* decW2p = A16(256 * 16);
  float* x     = AF(256 * 32);  // zero-padded to Kp=32
  float* pred  = AF(256 * 32);  // zero-padded to Kp=32
  float* h     = AF(65536);
  float* hn    = AF(65536);
  float* t1    = AF(65536);
  float* hu    = AF(131072);    // [delta_self | delta_inter], ld = 512
  float* hun   = AF(131072);
  float* rth   = AF(2097152);   // rt hidden, 8192 x 256
  float* logit = AF(131072);    // 8192 x 16 (8 valid)
  float* wts   = AF(65536);     // 8192 x 8

  auto pack = [&](const float* W, _Float16* Wp, int K, int N, int Kp, int Np,
                  int batch, int sS, int dS) {
    int total = batch * Kp * Np;
    k_pack<<<dim3((total + 255) / 256), dim3(256), 0, stream>>>(
        W, Wp, K, N, Kp, Np, batch, sS, dS);
  };
  auto gemm = [&](const float* A, int lda, int Kp, const _Float16* Wp,
                  const float* bias, float* C, int ldc, int Nreal, int Np,
                  int M, int relu) {
    if ((Np & 63) == 0)
      k_gemm<4><<<dim3(M / 16, Np / 64), dim3(32), 0, stream>>>(
          A, lda, Kp, Wp, bias, C, ldc, Nreal, relu);
    else
      k_gemm<1><<<dim3(M / 16, Np / 16), dim3(32), 0, stream>>>(
          A, lda, Kp, Wp, bias, C, ldc, Nreal, relu);
  };

  // ---- pack all weights to WMMA B layout (f16, padded) ----
  pack(enc_W1, encW1p, 8, 256, 32, 256, 1, 8 * 256, 32 * 256);
  pack(enc_W2, encW2p, 256, 256, 256, 256, 1, 65536, 65536);
  pack(rt_W1, rtW1p, 1024, 256, 1024, 256, 1, 262144, 262144);
  pack(rt_W2, rtW2p, 256, 8, 256, 16, 1, 2048, 4096);
  pack(fs_W1, fsW1p, 256, 256, 256, 256, 1, 65536, 65536);
  pack(fs_W2, fsW2p, 256, 256, 256, 256, 1, 65536, 65536);
  pack(ex_W1, exW1p, 512, 256, 512, 256, 8, 512 * 256, 512 * 256);
  pack(ex_W2, exW2p, 256, 256, 256, 256, 8, 256 * 256, 256 * 256);
  pack(up_W1, upW1p, 512, 256, 512, 256, 1, 131072, 131072);
  pack(up_W2, upW2p, 256, 256, 256, 256, 1, 65536, 65536);
  pack(dec_W1, decW1p, 256, 256, 256, 256, 1, 65536, 65536);
  pack(dec_W2, decW2p, 256, 8, 256, 16, 1, 2048, 4096);

  // ---- h = enc(gt_states[:,0]) ----
  k_zero<<<32, 256, 0, stream>>>(x, 8192);
  k_zero<<<32, 256, 0, stream>>>(pred, 8192);
  k_gather_x0<<<8, 256, 0, stream>>>(gt, x);
  gemm(x, 32, 32, encW1p, enc_b1, t1, 256, 256, 256, 256, 1);
  gemm(t1, 256, 256, encW2p, enc_b2, h, 256, 256, 256, 256, 0);

  for (int s = 0; s < 2; ++s) {
    // delta_self = fs MLP of LN(h) -> left half of hu
    k_ln<<<256, 256, 0, stream>>>(h, fs_g, fs_b, hn, 256);
    gemm(hn, 256, 256, fsW1p, fs_b1, t1, 256, 256, 256, 256, 1);
    k_zero<<<512, 256, 0, stream>>>(hu, 131072);
    gemm(t1, 256, 256, fsW2p, fs_b2, hu, 512, 256, 256, 256, 0);
    // relation weights: rt MLP over on-the-fly pair features + softmax
    k_gemm_pair4<<<dim3(512, 4), dim3(32), 0, stream>>>(h, rtW1p, rt_b1,
                                                        rth, 256);
    gemm(rth, 256, 256, rtW2p, rt_b2, logit, 16, 8, 16, 8192, 0);
    k_softmax8<<<32, 256, 0, stream>>>(logit, wts, 8192);
    // fused experts -> right half of hu (delta_inter)
    k_experts<<<512, 128, 0, stream>>>(h, exW1p, ex_b1, exW2p, ex_b2, wts,
                                       hu + 256, 512);
    // h = up MLP of LN(hu)
    k_ln<<<256, 256, 0, stream>>>(hu, up_g, up_b, hun, 512);
    gemm(hun, 512, 512, upW1p, up_b1, t1, 256, 256, 256, 256, 1);
    gemm(t1, 256, 256, upW2p, up_b2, h, 256, 256, 256, 256, 0);
    // pred = dec MLP(h)  (pred cols 8..31 stay zero for re-encode)
    gemm(h, 256, 256, decW1p, dec_b1, t1, 256, 256, 256, 256, 1);
    gemm(t1, 256, 256, decW2p, dec_b2, pred, 32, 8, 16, 256, 0);
    k_scatter_pred<<<8, 256, 0, stream>>>(pred, (float*)d_out, s);
    // h = enc(pred)
    gemm(pred, 32, 32, encW1p, enc_b1, t1, 256, 256, 256, 256, 1);
    gemm(t1, 256, 256, encW2p, enc_b2, h, 256, 256, 256, 256, 0);
  }
  k_copy_targets<<<16, 256, 0, stream>>>(gt, (float*)d_out + 4096);
}
